// Scattering_86260123174499
// MI455X (gfx1250) — matrix operand order repaired and checked by reading, not compiled, not run on hardware
//
#include <hip/hip_runtime.h>
#include <hip/hip_bf16.h>

typedef __attribute__((ext_vector_type(16))) _Float16 v16h;
typedef __attribute__((ext_vector_type(8)))  _Float16 v8h;
typedef __attribute__((ext_vector_type(2)))  _Float16 v2h;
typedef __attribute__((ext_vector_type(8)))  float    v8f;
typedef __attribute__((ext_vector_type(8)))  int      v8i;
typedef __attribute__((ext_vector_type(4)))  int      v4i;

#define EPS 1e-7f
#define NSAMP 524288
#define NCH   42
#define NCON  8
#define ZROW  1344                             // index of the all-zero A row
static const long NCTOT = (long)NSAMP * NCH;   // 22,020,096

#if __has_builtin(__builtin_amdgcn_rcpf)
#define RCPF(x) __builtin_amdgcn_rcpf(x)
#else
#define RCPF(x) (1.0f / (x))
#endif

#if __has_builtin(__builtin_amdgcn_wmma_f16_16x16x32_f16)
#define HAVE_F16ACC 1
#else
#define HAVE_F16ACC 0
#endif

// ---------------------------------------------------------------------------
// Kernel 1: tau_total sum + direct/diffuse transmittance (pure bandwidth)
// ---------------------------------------------------------------------------
__global__ void __launch_bounds__(256) tau_exp_kernel(
    const float4* __restrict__ tau4, const float* __restrict__ mud,
    const float* __restrict__ muf, float* __restrict__ out_td,
    float* __restrict__ out_tf)
{
    long i = (long)blockIdx.x * blockDim.x + threadIdx.x;   // flat (n, c)
    int  n = (int)(i / NCH);
    float4 a = tau4[i * 2 + 0];
    float4 b = tau4[i * 2 + 1];
    float s  = (a.x + a.y) + (a.z + a.w) + (b.x + b.y) + (b.z + b.w);
    float td = __expf(-s * RCPF(mud[n] + EPS));
    float tf = __expf(-s * RCPF(muf[n] + EPS));
    out_td[i] = td;
    out_tf[i] = tf;
}

// ---------------------------------------------------------------------------
// Kernel 2: batched per-channel MLP + softmax via WMMA.
// Hidden layers use f16-accumulate WMMA: the f16 C/D layout (rows 2v,2v+1 in
// VGPR v halves, rows 8-15 zero in upper lanes) is bit-identical to the next
// layer's f16 B layout, so a transition is just a packed ReLU + bias-1 slot.
// Biases are folded into the matmul via an augmented weight column.
// ---------------------------------------------------------------------------
__device__ __forceinline__ v16h swap16(v16h x) {
    v8i u = __builtin_bit_cast(v8i, x);
#pragma unroll
    for (int i = 0; i < 8; ++i)
        u[i] = __builtin_amdgcn_ds_swizzle(u[i], 0x401F);   // SWAPX16 (xor 16)
    return __builtin_bit_cast(v16h, u);
}

__device__ __forceinline__ v8h relu8h(v8h d) {
#if __has_builtin(__builtin_elementwise_max)
    v8h z = {};
    return __builtin_elementwise_max(d, z);   // v_pk_max_num_f16 x4
#else
    v8h r;
#pragma unroll
    for (int e = 0; e < 8; ++e)
        r[e] = (d[e] > (_Float16)0.0f) ? d[e] : (_Float16)0.0f;
    return r;
#endif
}

// f16 D -> next-layer f16 B: lower 4 VGPRs = relu(D), upper 4 = 0, plus the
// constant-1 slot feeding the next layer's bias column (K = slot).
__device__ __forceinline__ v16h packB16(v8h d, int slot, bool lo) {
    v4i ri = __builtin_bit_cast(v4i, relu8h(d));
    v8i hi = {ri[0], ri[1], ri[2], ri[3], 0, 0, 0, 0};
    v16h h = __builtin_bit_cast(v16h, hi);
    h[slot] = lo ? (_Float16)1.0f : (_Float16)0.0f;
    return h;
}

__device__ __forceinline__ v16h relupack32(v8f d, int slot, bool lo) {
    v16h h = {};
#pragma unroll
    for (int e = 0; e < 8; ++e)
        h[e] = (_Float16)fmaxf(d[e], 0.0f);
    h[slot] = lo ? (_Float16)1.0f : (_Float16)0.0f;
    return h;
}

__device__ __forceinline__ void softmax3(v8f d, float* p) {
    float l0 = d[0], l1 = d[1], l2 = d[2];      // bias already in the matmul
    float m  = fmaxf(l0, fmaxf(l1, l2));
    float e0 = __expf(l0 - m), e1 = __expf(l1 - m), e2 = __expf(l2 - m);
    float inv = RCPF(e0 + e1 + e2);
    p[0] = e0 * inv; p[1] = e1 * inv; p[2] = e2 * inv;
}

// NET = 0 (direct, in0=9+bias via upper half), NET = 1 (diffuse, in0=8, bias upper)
template <int NET>
__device__ __forceinline__ void mlp_net(const _Float16* sW, const int* sU,
                                        int c, int lane, v16h h0, v16h h1,
                                        float* __restrict__ outp, long oidx)
{
    const bool lo = lane < 16;
    const int outs[4] = {5, 4, 4, 3};
    const int rb[2][4] = {{0, 210, 378, 546}, {672, 882, 1050, 1218}};
    const v8f z = {};
#pragma unroll
    for (int i = 0; i < 4; ++i) {
        // ---- A = augmented weight tile; invalid lanes read the zero row ----
        int rowIdx = (lane < outs[i]) ? (rb[NET][i] + c * outs[i] + lane) : ZROW;
        v16h a = *(const v16h*)&sW[rowIdx * 16];
        if (i == 0) {   // lower-half elements 8..9 (K=8..9): (mu-col, bias) / (bias, 0)
            int ur = lane;
            int uc = (ur >= 0 && ur < 5) ? ur : 0;
            int u  = sU[(NET * NCH + c) * 5 + uc];
            v8i ai = __builtin_bit_cast(v8i, a);
            ai[4]  = (lane < 5) ? u : ai[4];
            a      = __builtin_bit_cast(v16h, ai);
        }

        if (i < 3) {
#if HAVE_F16ACC
            v8h z8 = {};
            v8h d0 = __builtin_amdgcn_wmma_f16_16x16x32_f16(false, a, false, h0,
                                                            (short)0, z8, false, false);
            v8h d1 = __builtin_amdgcn_wmma_f16_16x16x32_f16(false, a, false, h1,
                                                            (short)0, z8, false, false);
            h0 = packB16(d0, outs[i], lo);
            h1 = packB16(d1, outs[i], lo);
#else
            v8f d0 = __builtin_amdgcn_wmma_f32_16x16x32_f16(false, a, false, h0,
                                                            (short)0, z, false, false);
            v8f d1 = __builtin_amdgcn_wmma_f32_16x16x32_f16(false, a, false, h1,
                                                            (short)0, z, false, false);
            h0 = relupack32(d0, outs[i], lo);
            h1 = relupack32(d1, outs[i], lo);
#endif
        } else {        // final layer: f32 logits for softmax
            v8f d0 = __builtin_amdgcn_wmma_f32_16x16x32_f16(false, a, false, h0,
                                                            (short)0, z, false, false);
            v8f d1 = __builtin_amdgcn_wmma_f32_16x16x32_f16(false, a, false, h1,
                                                            (short)0, z, false, false);
            float p0[3], p1[3];
            softmax3(d0, p0);
            softmax3(d1, p1);
#pragma unroll
            for (int o = 0; o < 3; ++o) {   // lanes 16-31 emit tile1's samples
                int   sh = __builtin_amdgcn_ds_swizzle(__float_as_int(p1[o]), 0x401F);
                float v  = lo ? p0[o] : __int_as_float(sh);
                outp[oidx + o] = v;
            }
        }
    }
}

// Stage W (and bias column when in < 8) as 16-half A-register row images.
__device__ __forceinline__ void fillW(_Float16* sW, const float* __restrict__ W,
                                      const float* __restrict__ b,
                                      int base, int outc, int in, int tid)
{
    const int in8 = (in < 8) ? in : 8;
    int tot = NCH * outc * 16;
    for (int idx = tid; idx < tot; idx += 256) {
        int k = idx & 15, row = idx >> 4;
        int c = row / outc, r = row - c * outc;
        float v = 0.0f;
        if (k < in8)                 v = W[(c * outc + r) * in + k];
        else if (in < 8 && k == in)  v = b[c * outc + r];   // augmented bias col
        sW[(base + row) * 16 + k] = (_Float16)v;
    }
}

__global__ void __launch_bounds__(256) mlp_wmma_kernel(
    const float* __restrict__ cons, const float* __restrict__ mud,
    const float* __restrict__ Wd0, const float* __restrict__ bd0,
    const float* __restrict__ Wd1, const float* __restrict__ bd1,
    const float* __restrict__ Wd2, const float* __restrict__ bd2,
    const float* __restrict__ Wd3, const float* __restrict__ bd3,
    const float* __restrict__ Wf0, const float* __restrict__ bf0,
    const float* __restrict__ Wf1, const float* __restrict__ bf1,
    const float* __restrict__ Wf2, const float* __restrict__ bf2,
    const float* __restrict__ Wf3, const float* __restrict__ bf3,
    float* __restrict__ out)
{
    __shared__ __align__(32) _Float16 sW[(ZROW + 1) * 16];  // 43040 B + zero row
    __shared__ int sU[2 * NCH * 5];                         // layer0 upper: {K8,K9}

    const int tid  = threadIdx.x;
    const int lane = tid & 31;
    const int wave = tid >> 5;

    // ---- stage all (augmented) weights in LDS once per block ----
    fillW(sW, Wd0, bd0,    0, 5, 9, tid);
    fillW(sW, Wd1, bd1,  210, 4, 5, tid);
    fillW(sW, Wd2, bd2,  378, 4, 4, tid);
    fillW(sW, Wd3, bd3,  546, 3, 4, tid);
    fillW(sW, Wf0, bf0,  672, 5, 8, tid);
    fillW(sW, Wf1, bf1,  882, 4, 5, tid);
    fillW(sW, Wf2, bf2, 1050, 4, 4, tid);
    fillW(sW, Wf3, bf3, 1218, 3, 4, tid);
    if (tid < 16) sW[ZROW * 16 + tid] = (_Float16)0.0f;     // the zero row
    for (int idx = tid; idx < 2 * NCH * 5; idx += 256) {
        int net = idx / (NCH * 5);
        int rem = idx - net * NCH * 5;
        int c = rem / 5, r = rem - c * 5;
        v2h p;
        if (net == 0) {  // direct: K=8 mu column, K=9 bias
            p[0] = (_Float16)Wd0[(c * 5 + r) * 9 + 8];
            p[1] = (_Float16)bd0[c * 5 + r];
        } else {         // diffuse: K=8 bias
            p[0] = (_Float16)bf0[c * 5 + r];
            p[1] = (_Float16)0.0f;
        }
        sU[idx] = __builtin_bit_cast(int, p);
    }
    __syncthreads();

    // ---- per-lane sample inputs: wave handles 32 samples = 2 WMMA tiles ----
    const int n  = (blockIdx.x * 8 + wave) * 32 + lane;
    const bool lo = lane < 16;

    float mu   = mud[n];
    float invm = RCPF(mu + EPS);
    const float4* c4 = (const float4*)(cons + (long)n * NCON);
    float4 x0 = c4[0], x1 = c4[1];
    float x[8] = {x0.x, x0.y, x0.z, x0.w, x1.x, x1.y, x1.z, x1.w};

    v16h md = {}, mf = {};
#pragma unroll
    for (int k = 0; k < 8; ++k) {
        md[k] = (_Float16)(x[k] * invm);
        mf[k] = (_Float16)x[k];
    }
    md[8] = (_Float16)mu;       // direct feature 8 = mu (K=8)
    md[9] = (_Float16)1.0f;     // direct bias slot  (K=9)
    mf[8] = (_Float16)1.0f;     // diffuse bias slot (K=8)

    const v16h zv = {};
    v16h mdS = swap16(md), mfS = swap16(mf);
    v16h Bd0 = lo ? md  : zv;   // tile0: samples base..base+15
    v16h Bd1 = lo ? mdS : zv;   // tile1: samples base+16..base+31
    v16h Bf0 = lo ? mf  : zv;
    v16h Bf1 = lo ? mfS : zv;

    const long obase_d = 2L * NCTOT + (long)n * NCH * 3;
    const long obase_f = 5L * NCTOT + (long)n * NCH * 3;

    for (int c = 0; c < NCH; ++c) {
        mlp_net<0>(sW, sU, c, lane, Bd0, Bd1, out, obase_d + c * 3);
        mlp_net<1>(sW, sU, c, lane, Bf0, Bf1, out, obase_f + c * 3);
    }
}

// ---------------------------------------------------------------------------
extern "C" void kernel_launch(void* const* d_in, const int* in_sizes, int n_in,
                              void* d_out, int out_size, void* d_ws, size_t ws_size,
                              hipStream_t stream)
{
    const float* tau  = (const float*)d_in[0];
    const float* mud  = (const float*)d_in[1];
    const float* muf  = (const float*)d_in[2];
    const float* cons = (const float*)d_in[3];
    const float* Wd0 = (const float*)d_in[4],  *bd0 = (const float*)d_in[5];
    const float* Wf0 = (const float*)d_in[6],  *bf0 = (const float*)d_in[7];
    const float* Wd1 = (const float*)d_in[8],  *bd1 = (const float*)d_in[9];
    const float* Wf1 = (const float*)d_in[10], *bf1 = (const float*)d_in[11];
    const float* Wd2 = (const float*)d_in[12], *bd2 = (const float*)d_in[13];
    const float* Wf2 = (const float*)d_in[14], *bf2 = (const float*)d_in[15];
    const float* Wd3 = (const float*)d_in[16], *bd3 = (const float*)d_in[17];
    const float* Wf3 = (const float*)d_in[18], *bf3 = (const float*)d_in[19];
    float* out = (float*)d_out;

    // t_direct / t_diffuse: one thread per (n, c)
    int blocks1 = (int)(NCTOT / 256);     // 86016
    tau_exp_kernel<<<blocks1, 256, 0, stream>>>(
        (const float4*)tau, mud, muf, out, out + NCTOT);

    // MLP softmax splits: 256 samples per block (8 waves x 32 samples)
    int blocks2 = NSAMP / 256;            // 2048
    mlp_wmma_kernel<<<blocks2, 256, 0, stream>>>(
        cons, mud,
        Wd0, bd0, Wd1, bd1, Wd2, bd2, Wd3, bd3,
        Wf0, bf0, Wf1, bf1, Wf2, bf2, Wf3, bf3,
        out);
}
